// EngramModule_17592186045041
// MI455X (gfx1250) — matrix operand order repaired and checked by reading, not compiled, not run on hardware
//
#include <hip/hip_runtime.h>
#include <stdint.h>

// ---------------- problem constants ----------------
#define BATCH   8
#define TSEQ    2048
#define NEMBD   2048
#define DMEM    1024          // (MAX_NGRAM-1)*N_HEADS*EMBED_DIM
#define NTOK    (BATCH*TSEQ)  // 16384
#define NTAB    16            // (3-1)*8 hash tables
#define EDIM    64

typedef __attribute__((ext_vector_type(16))) __bf16 v16bf;
typedef __attribute__((ext_vector_type(8)))  __bf16 v8bf;
typedef __attribute__((ext_vector_type(8)))  float  v8f;

union ABfrag { v16bf v; v8bf h[2]; };

// ---------------- compile-time hash constants ----------------
struct HashConsts {
    unsigned p[NTAB];
    unsigned pw[NTAB][3];
    int      n[NTAB];
    unsigned off[NTAB];
};

static constexpr bool odd_is_prime(unsigned n) {
    for (unsigned d = 3; d * d <= n; d += 2)
        if (n % d == 0) return false;
    return true;
}
static constexpr unsigned next_prime(unsigned n) {
    if (n <= 2) return 2;
    if (n % 2 == 0) n += 1;
    while (!odd_is_prime(n)) n += 2;
    return n;
}
static constexpr HashConsts make_hash_consts() {
    HashConsts h{};
    unsigned acc = 0;
    int i = 0;
    for (int n = 2; n <= 3; ++n) {
        for (int k = 0; k < 8; ++k, ++i) {
            unsigned p = next_prime(98304u + (unsigned)(n * k * 997));
            unsigned long long seed =
                (unsigned long long)n * 2654435761ull + (unsigned long long)k * 40503ull;
            unsigned s = (unsigned)(seed % p);
            for (int j = 0; j < n; ++j) {
                int e = n - j;
                unsigned long long r = 1;
                for (int t = 0; t < e; ++t) r = (r * s) % p;
                h.pw[i][j] = (unsigned)r;
            }
            h.p[i] = p; h.n[i] = n; h.off[i] = acc;
            acc += p;
        }
    }
    return h;
}
__constant__ HashConsts g_hc = make_hash_consts();

// ---------------- helpers ----------------
__device__ __forceinline__ unsigned short f2bf(float f) {
    unsigned u = __float_as_uint(f);
    unsigned r = u + 0x7FFFu + ((u >> 16) & 1u);   // RNE
    return (unsigned short)(r >> 16);
}

// ---------------- kernel 0: f32 -> bf16 weight staging ----------------
__global__ void cvt_bf16_kernel(const float* __restrict__ src,
                                unsigned short* __restrict__ dst, int n4) {
    int i = blockIdx.x * blockDim.x + threadIdx.x;
    if (i >= n4) return;
    float4 v = ((const float4*)src)[i];
    unsigned long long pk =
        (unsigned long long)f2bf(v.x)
      | ((unsigned long long)f2bf(v.y) << 16)
      | ((unsigned long long)f2bf(v.z) << 32)
      | ((unsigned long long)f2bf(v.w) << 48);
    ((unsigned long long*)dst)[i] = pk;
}

// ---------------- kernel 1: n-gram hash + table gather -> e (bf16) ----------
// one block per token; thread handles 4 consecutive components of e
__global__ void hash_gather_kernel(const int* __restrict__ ids,
                                   const float* __restrict__ tables,
                                   unsigned short* __restrict__ e_out) {
    const int token = blockIdx.x;
    const int b = token >> 11, t = token & (TSEQ - 1);
    const int m = threadIdx.x * 4;          // element of e row (0..1023)
    const int i = m >> 6;                   // table id
    const int c = m & 63;                   // component in 64-dim row

    const int      n = g_hc.n[i];
    const unsigned p = g_hc.p[i];
    unsigned acc = 0;
#pragma unroll
    for (int j = 0; j < 3; ++j) {
        if (j < n) {
            int tp = t - (n - 1) + j;
            unsigned idv = (tp >= 0) ? (unsigned)ids[b * TSEQ + tp] : 0u;
            acc = (acc + (idv * g_hc.pw[i][j]) % p) % p;   // u32 exact: product < 2^32
        }
    }
    size_t row = ((size_t)(g_hc.off[i] + acc)) * EDIM + c;
    float4 v = *(const float4*)(tables + row);
    unsigned long long pk =
        (unsigned long long)f2bf(v.x)
      | ((unsigned long long)f2bf(v.y) << 16)
      | ((unsigned long long)f2bf(v.z) << 32)
      | ((unsigned long long)f2bf(v.w) << 48);
    *(unsigned long long*)(e_out + (size_t)token * DMEM + m) = pk;
}

// ---------------- kernel 2/4: bf16 WMMA GEMM  C[M,N] = E[M,K] * W[N,K]^T ----
// M=16384 tokens, N=2048, K=1024. Each wave: 32x64 output (2 M-tiles x 4
// N-tiles, 8 v8f accumulators = 64 VGPRs -> no spills). Per K-step: 8 B-loads
// shared across 2 M-tiles + 4 A-loads -> 12 b128 loads for 8 WMMAs.
// All sub-tile displacements are compile-time immediates; a single A and a
// single B pointer advance by 32 elements per K-step.
template <bool GATED>
__global__ __launch_bounds__(256, 1)
void gemm_wmma_kernel(const unsigned short* __restrict__ Ebf,
                      const unsigned short* __restrict__ Wbf,
                      const float* __restrict__ gate,
                      float* __restrict__ out) {
    const int lane = threadIdx.x & 31;
    const int wave = threadIdx.x >> 5;
    const int tile = blockIdx.x * 8 + wave;      // 16384 waves total
    const int mGrp = tile & 511;                 // 512 M-groups of 32 rows
    const int nGrp = tile >> 9;                  // 32 N-groups of 64 cols
    const int m0 = mGrp * 32;
    const int n0 = nGrp * 64;

    const __bf16* E = reinterpret_cast<const __bf16*>(Ebf);
    const __bf16* W = reinterpret_cast<const __bf16*>(Wbf);

    // A fragment addressing (16-bit A 16x32, ISA p.108):
    //   lanes 0-15: row=lane,    halves at K {kb..kb+7, kb+16..kb+23}, kb=0
    //   lanes16-31: row=lane-16, kb=8
    const int akb = (lane >> 4) * 8;
    const __bf16* aPtr = E + (size_t)(m0 + (lane & 15)) * DMEM + akb;

    // B fragment addressing (16-bit B 32x16): lane col = lane&15,
    //   halves at K {kb..kb+7, kb+16..kb+23}, kb=(lane>>4)*8 (matches A).
    const int bkb = (lane >> 4) * 8;
    const __bf16* bPtr = W + (size_t)(n0 + (lane & 15)) * DMEM + bkb;

    v8f acc[2][4] = {};

    for (int kk = 0; kk < DMEM / 32; ++kk) {
        ABfrag bf[4];
#pragma unroll
        for (int s = 0; s < 4; ++s) {
            bf[s].h[0] = *(const v8bf*)(bPtr + s * 16 * DMEM);
            bf[s].h[1] = *(const v8bf*)(bPtr + s * 16 * DMEM + 16);
        }
#pragma unroll
        for (int mt = 0; mt < 2; ++mt) {
            ABfrag a;
            a.h[0] = *(const v8bf*)(aPtr + mt * 16 * DMEM);
            a.h[1] = *(const v8bf*)(aPtr + mt * 16 * DMEM + 16);
#pragma unroll
            for (int s = 0; s < 4; ++s) {
                acc[mt][s] = __builtin_amdgcn_wmma_f32_16x16x32_bf16(
                    false, a.v, false, bf[s].v, (short)0, acc[mt][s], false, false);
            }
        }
        aPtr += 32;
        bPtr += 32;
    }

    // C layout: lane col = lane&15, rows = (lane>>4)*8 + r
#pragma unroll
    for (int mt = 0; mt < 2; ++mt) {
        const int rbase = m0 + mt * 16 + (lane >> 4) * 8;
        float g[8];
        if (GATED) {
#pragma unroll
            for (int r = 0; r < 8; ++r) g[r] = gate[rbase + r];
        }
#pragma unroll
        for (int s = 0; s < 4; ++s) {
            const int col = n0 + s * 16 + (lane & 15);
#pragma unroll
            for (int r = 0; r < 8; ++r) {
                float v = acc[mt][s][r];
                if (GATED) v *= g[r];
                out[(size_t)(rbase + r) * NEMBD + col] = v;
            }
        }
    }
}

// ---------------- reductions ----------------
__device__ __forceinline__ float wave_sum(float v) {
#pragma unroll
    for (int off = 16; off > 0; off >>= 1) v += __shfl_xor(v, off, 32);
    return v;
}

// kernel 3: per-token gate = sigmoid( dot(rms(x), rms(kpre)) / sqrt(D) )
__global__ void gate_kernel(const float* __restrict__ x,
                            const float* __restrict__ kpre,
                            float* __restrict__ gate) {
    const int token = blockIdx.x;
    const float* xr = x    + (size_t)token * NEMBD;
    const float* kr = kpre + (size_t)token * NEMBD;
    float sxx = 0.f, skk = 0.f, sxk = 0.f;
    for (int d = threadIdx.x; d < NEMBD; d += 256) {
        float xv = xr[d], kv = kr[d];
        sxx = fmaf(xv, xv, sxx);
        skk = fmaf(kv, kv, skk);
        sxk = fmaf(xv, kv, sxk);
    }
    sxx = wave_sum(sxx); skk = wave_sum(skk); sxk = wave_sum(sxk);
    __shared__ float s0[8], s1[8], s2[8];
    const int wv = threadIdx.x >> 5, ln = threadIdx.x & 31;
    if (ln == 0) { s0[wv] = sxx; s1[wv] = skk; s2[wv] = sxk; }
    __syncthreads();
    if (threadIdx.x == 0) {
        float a = 0.f, b = 0.f, c = 0.f;
#pragma unroll
        for (int i = 0; i < 8; ++i) { a += s0[i]; b += s1[i]; c += s2[i]; }
        const float eps = 1.1920929e-7f;
        float mx = a * (1.f / NEMBD) + eps;
        float mk = b * (1.f / NEMBD) + eps;
        float z = c * rsqrtf(mx) * rsqrtf(mk) * 0.022097086912079608f; // 1/sqrt(2048)
        gate[token] = 1.f / (1.f + expf(-z));
    }
}

// kernel 5: per-token inverse rms of gated_v
__global__ void rscale_kernel(const float* __restrict__ gv,
                              float* __restrict__ rscale) {
    const int token = blockIdx.x;
    const float* gr = gv + (size_t)token * NEMBD;
    float ss = 0.f;
    for (int d = threadIdx.x; d < NEMBD; d += 256) {
        float v = gr[d];
        ss = fmaf(v, v, ss);
    }
    ss = wave_sum(ss);
    __shared__ float s0[8];
    const int wv = threadIdx.x >> 5, ln = threadIdx.x & 31;
    if (ln == 0) s0[wv] = ss;
    __syncthreads();
    if (threadIdx.x == 0) {
        float a = 0.f;
#pragma unroll
        for (int i = 0; i < 8; ++i) a += s0[i];
        rscale[token] = rsqrtf(a * (1.f / NEMBD) + 1.1920929e-7f);
    }
}

// kernel 6: depthwise dilated causal conv (K=4, dil=3) on normed_v, silu, +gated_v
__global__ void conv_silu_kernel(const float* __restrict__ gv,
                                 const float* __restrict__ rscale,
                                 const float* __restrict__ conv_w,
                                 float* __restrict__ out) {
    const int idx = blockIdx.x * blockDim.x + threadIdx.x;   // NTOK*512 threads
    const int token = idx >> 9;
    const int d4 = (idx & 511) * 4;
    const int b = token >> 11, t = token & (TSEQ - 1);

    float ax = 0.f, ay = 0.f, az = 0.f, aw = 0.f;
#pragma unroll
    for (int j = 0; j < 4; ++j) {
        int tj = t - 9 + 3 * j;
        if (tj >= 0) {
            int tok_j = b * TSEQ + tj;
            float rs = rscale[tok_j];
            float4 g = *(const float4*)(gv + (size_t)tok_j * NEMBD + d4);
            ax = fmaf(g.x * rs, conv_w[(d4 + 0) * 4 + j], ax);
            ay = fmaf(g.y * rs, conv_w[(d4 + 1) * 4 + j], ay);
            az = fmaf(g.z * rs, conv_w[(d4 + 2) * 4 + j], az);
            aw = fmaf(g.w * rs, conv_w[(d4 + 3) * 4 + j], aw);
        }
    }
    float4 gt = *(const float4*)(gv + (size_t)token * NEMBD + d4);
    float4 o;
    o.x = ax / (1.f + expf(-ax)) + gt.x;
    o.y = ay / (1.f + expf(-ay)) + gt.y;
    o.z = az / (1.f + expf(-az)) + gt.z;
    o.w = aw / (1.f + expf(-aw)) + gt.w;
    *(float4*)(out + (size_t)token * NEMBD + d4) = o;
}

// ---------------- workspace layout ----------------
static constexpr size_t OFF_E    = 0;                                    // e  bf16: 32 MiB
static constexpr size_t OFF_KW   = OFF_E  + (size_t)NTOK * DMEM * 2;     // key_w bf16: 4 MiB
static constexpr size_t OFF_VW   = OFF_KW + (size_t)NEMBD * DMEM * 2;    // value_w bf16: 4 MiB
static constexpr size_t OFF_KP   = OFF_VW + (size_t)NEMBD * DMEM * 2;    // kpre (later gated_v): 128 MiB
static constexpr size_t OFF_GATE = OFF_KP + (size_t)NTOK * NEMBD * 4;    // 64 KiB
static constexpr size_t OFF_RS   = OFF_GATE + (size_t)NTOK * 4;          // 64 KiB

extern "C" void kernel_launch(void* const* d_in, const int* in_sizes, int n_in,
                              void* d_out, int out_size, void* d_ws, size_t ws_size,
                              hipStream_t stream) {
    (void)in_sizes; (void)n_in; (void)out_size; (void)ws_size;

    const float* x       = (const float*)d_in[0];
    const int*   ids     = (const int*)  d_in[1];
    const float* tables  = (const float*)d_in[2];
    const float* key_w   = (const float*)d_in[3];
    const float* value_w = (const float*)d_in[4];
    const float* conv_w  = (const float*)d_in[5];
    float* out = (float*)d_out;

    char* ws = (char*)d_ws;
    unsigned short* eB  = (unsigned short*)(ws + OFF_E);
    unsigned short* kwB = (unsigned short*)(ws + OFF_KW);
    unsigned short* vwB = (unsigned short*)(ws + OFF_VW);
    float* kpre   = (float*)(ws + OFF_KP);   // aliased as gated_v after gate_kernel
    float* gv     = kpre;
    float* gate   = (float*)(ws + OFF_GATE);
    float* rscale = (float*)(ws + OFF_RS);

    // 0. stage weights as bf16  (2048*1024 elems each -> 524288 float4s)
    cvt_bf16_kernel<<<2048, 256, 0, stream>>>(key_w,   kwB, NEMBD * DMEM / 4);
    cvt_bf16_kernel<<<2048, 256, 0, stream>>>(value_w, vwB, NEMBD * DMEM / 4);

    // 1. hash + gather -> e (bf16)
    hash_gather_kernel<<<NTOK, 256, 0, stream>>>(ids, tables, eB);

    // 2. kpre = e @ key_w^T   (WMMA bf16)  -- 16384 waves, 8 per block
    gemm_wmma_kernel<false><<<2048, 256, 0, stream>>>(eB, kwB, nullptr, kpre);

    // 3. gate per token
    gate_kernel<<<NTOK, 256, 0, stream>>>(x, kpre, gate);

    // 4. gated_v = gate * (e @ value_w^T)   (WMMA bf16, overwrites kpre buffer)
    gemm_wmma_kernel<true><<<2048, 256, 0, stream>>>(eB, vwB, gate, gv);

    // 5. inverse rms of gated_v
    rscale_kernel<<<NTOK, 256, 0, stream>>>(gv, rscale);

    // 6. dilated depthwise conv + silu + residual
    conv_silu_kernel<<<NTOK * 512 / 256, 256, 0, stream>>>(gv, rscale, conv_w, out);
}
